// Pred_endecoder_53901839565220
// MI455X (gfx1250) — compile-verified, matrix operand order checked by reading, not run on hardware
//
#include <hip/hip_runtime.h>
#include <math.h>

// ---------------------------------------------------------------------------
// transformer2 Pred_endecoder trans_block for MI455X (gfx1250, wave32, WMMA)
//
// fp32 activations, f16 WMMA (v_wmma_f32_16x16x32_f16) with f32 accumulate.
// GEMM: 128x128 block tile, 8 waves each owning 32x64 (8 v8f accumulators),
// double-buffered LDS staging via GLOBAL_LOAD_ASYNC_TO_LDS_B128 (ASYNCcnt).
// ---------------------------------------------------------------------------

typedef _Float16 half_t;
typedef __attribute__((ext_vector_type(16))) _Float16 v16h;
typedef __attribute__((ext_vector_type(8)))  _Float16 v8h;
typedef __attribute__((ext_vector_type(8)))  float    v8f;

#define BB    32
#define CDIM  512
#define TDIM  256
#define NHD   4
#define DHD   128
#define EDIM  2048
#define HIN   44
#define WIN   44

// ---- CDNA5 async global->LDS copy helpers -------------------------------
__device__ __forceinline__ void async_load_b128(const half_t* lds_dst,
                                                const half_t* gsrc)
{
  // VDST carries the per-lane LDS byte offset (generic ptr low 32 bits ==
  // AS3 offset: the LDS aperture keeps the low 32 address bits).
  unsigned lds_off = (unsigned)(size_t)lds_dst;
  asm volatile("global_load_async_to_lds_b128 %0, %1, off"
               :: "v"(lds_off), "v"(gsrc) : "memory");
}

__device__ __forceinline__ void wait_async()
{
#if __has_builtin(__builtin_amdgcn_s_wait_asynccnt)
  __builtin_amdgcn_s_wait_asynccnt(0);
#else
  asm volatile("s_wait_asynccnt 0x0" ::: "memory");
#endif
}

// ---------------------------------------------------------------------------
// Bilinear 44x44 -> 16x16 (align_corners), token-major output + pos add.
// ---------------------------------------------------------------------------
__global__ __launch_bounds__(256) void resize_down_kernel(
    const float* __restrict__ in, const float* __restrict__ pos,
    float* __restrict__ outf, half_t* __restrict__ outh)
{
  long idx = (long)blockIdx.x * 256 + threadIdx.x;
  const long total = (long)BB * TDIM * CDIM;
  if (idx >= total) return;
  int c = (int)(idx % CDIM);
  int t = (int)((idx / CDIM) % TDIM);
  int b = (int)(idx / ((long)CDIM * TDIM));
  int oy = t >> 4, ox = t & 15;
  const float sc = (float)(HIN - 1) / 15.0f;
  float fy = oy * sc, fx = ox * sc;
  int y0 = (int)floorf(fy); if (y0 > HIN - 2) y0 = HIN - 2;
  int x0 = (int)floorf(fx); if (x0 > WIN - 2) x0 = WIN - 2;
  float wy = fy - (float)y0, wx = fx - (float)x0;
  const float* p = in + (long)(b * CDIM + c) * HIN * WIN;
  float a00 = p[y0 * WIN + x0],       a01 = p[y0 * WIN + x0 + 1];
  float a10 = p[(y0 + 1) * WIN + x0], a11 = p[(y0 + 1) * WIN + x0 + 1];
  float v0 = a00 + (a01 - a00) * wx;
  float v1 = a10 + (a11 - a10) * wx;
  float v  = v0 + (v1 - v0) * wy + pos[(long)t * CDIM + c];
  outf[idx] = v;
  outh[idx] = (half_t)v;
}

// ---------------------------------------------------------------------------
// Bilinear 16x16 -> 44x44 (align_corners) from token-major [B,T,C] f32.
// ---------------------------------------------------------------------------
__global__ __launch_bounds__(256) void resize_up_kernel(
    const float* __restrict__ tok, float* __restrict__ out)
{
  long idx = (long)blockIdx.x * 256 + threadIdx.x;
  const long total = (long)BB * CDIM * HIN * WIN;
  if (idx >= total) return;
  int ox = (int)(idx % WIN);
  int oy = (int)((idx / WIN) % HIN);
  int c  = (int)((idx / ((long)WIN * HIN)) % CDIM);
  int b  = (int)(idx / ((long)WIN * HIN * CDIM));
  const float sc = 15.0f / (float)(HIN - 1);
  float fy = oy * sc, fx = ox * sc;
  int y0 = (int)floorf(fy); if (y0 > 14) y0 = 14;
  int x0 = (int)floorf(fx); if (x0 > 14) x0 = 14;
  float wy = fy - (float)y0, wx = fx - (float)x0;
  const float* base = tok + (long)b * TDIM * CDIM + c;
  float a00 = base[(long)(y0 * 16 + x0) * CDIM];
  float a01 = base[(long)(y0 * 16 + x0 + 1) * CDIM];
  float a10 = base[(long)((y0 + 1) * 16 + x0) * CDIM];
  float a11 = base[(long)((y0 + 1) * 16 + x0 + 1) * CDIM];
  float v0 = a00 + (a01 - a00) * wx;
  float v1 = a10 + (a11 - a10) * wx;
  out[idx] = v0 + (v1 - v0) * wy;
}

// ---------------------------------------------------------------------------
// Weight prep: fp32 [K,N] (torch Linear layout, y = x@W) -> f16 [N,K].
// ---------------------------------------------------------------------------
__global__ __launch_bounds__(256) void wprep_kernel(
    const float* __restrict__ W, half_t* __restrict__ WT, int K, int N)
{
  long idx = (long)blockIdx.x * 256 + threadIdx.x;
  long total = (long)K * N;
  if (idx >= total) return;
  int k = (int)(idx % K);
  int n = (int)(idx / K);
  WT[(long)n * K + k] = (half_t)W[(long)k * N + n];
}

// ---------------------------------------------------------------------------
// LayerNorm over C=512, one row per 256-thread block. Optional f32/f16 out.
// ---------------------------------------------------------------------------
__global__ __launch_bounds__(256) void layernorm_kernel(
    const float* __restrict__ x, const float* __restrict__ g,
    const float* __restrict__ be, float* __restrict__ outf,
    half_t* __restrict__ outh)
{
  __shared__ float ps[8], ps2[8];
  long row = blockIdx.x;
  const float* xr = x + row * CDIM;
  float s = 0.f, s2 = 0.f;
  for (int i = threadIdx.x; i < CDIM; i += 256) {
    float v = xr[i]; s += v; s2 += v * v;
  }
  int lane = threadIdx.x & 31, wid = threadIdx.x >> 5;
  #pragma unroll
  for (int m = 16; m; m >>= 1) { s += __shfl_xor(s, m, 32); s2 += __shfl_xor(s2, m, 32); }
  if (lane == 0) { ps[wid] = s; ps2[wid] = s2; }
  __syncthreads();
  if (threadIdx.x == 0) {
    float a = 0.f, b2 = 0.f;
    for (int i = 0; i < 8; i++) { a += ps[i]; b2 += ps2[i]; }
    ps[0] = a; ps2[0] = b2;
  }
  __syncthreads();
  float mean = ps[0] * (1.0f / CDIM);
  float var  = ps2[0] * (1.0f / CDIM) - mean * mean;
  float rstd = rsqrtf(var + 1e-5f);
  for (int i = threadIdx.x; i < CDIM; i += 256) {
    float v = (xr[i] - mean) * rstd * g[i] + be[i];
    if (outf) outf[row * CDIM + i] = v;
    if (outh) outh[row * CDIM + i] = (half_t)v;
  }
}

// ---------------------------------------------------------------------------
// Softmax over rows of 256, one wave32 per row, folds 1/sqrt(Dh) scale.
// ---------------------------------------------------------------------------
__global__ __launch_bounds__(256) void softmax_kernel(
    const float* __restrict__ S, half_t* __restrict__ P, float scale)
{
  int lane = threadIdx.x & 31, wid = threadIdx.x >> 5;
  long row = (long)blockIdx.x * 8 + wid;
  const float* sr = S + row * TDIM;
  float v[8];
  float m = -1e30f;
  #pragma unroll
  for (int i = 0; i < 8; i++) { v[i] = sr[lane + i * 32] * scale; m = fmaxf(m, v[i]); }
  #pragma unroll
  for (int off = 16; off; off >>= 1) m = fmaxf(m, __shfl_xor(m, off, 32));
  float sum = 0.f;
  #pragma unroll
  for (int i = 0; i < 8; i++) { v[i] = __expf(v[i] - m); sum += v[i]; }
  #pragma unroll
  for (int off = 16; off; off >>= 1) sum += __shfl_xor(sum, off, 32);
  float inv = 1.0f / sum;
  #pragma unroll
  for (int i = 0; i < 8; i++) P[row * TDIM + lane + i * 32] = (half_t)(v[i] * inv);
}

// ---------------------------------------------------------------------------
// Batched WMMA GEMM: C = A(MxK,f16) * B + bias [+ residual] [relu]
//   BNT=true : B memory is [N,K]  -> async b128 LDS staging
//   BNT=false: B memory is [K,N]  -> transpose-staged manually (P*V only)
// Block 256 threads (8 waves), tile 128(M) x 128(N), K step 32, double-
// buffered LDS: chunk k+1 DMA'd (ASYNCcnt) while chunk k computes.
// Wave w: rows (w&3)*32..+31, cols (w>>2)*64..+63 => 2x4 v8f accumulators.
// ---------------------------------------------------------------------------
struct GemmParams {
  const half_t* A; long lda, sAb, sAh;
  const half_t* B; long ldb, sBb, sBh;
  float* Cf; half_t* Ch; long ldc, sCb, sCh;
  const float* bias;
  const float* R; long ldr, sRb, sRh;
  int M, N, K, NHb, relu;
};

#define LDSPITCH 40   // 32 halves + 8 pad: 80B row stride spreads LDS banks

template <bool BNT>
__global__ __launch_bounds__(256) void gemm_wmma_kernel(GemmParams p)
{
  __shared__ half_t lsA[2][128 * LDSPITCH];
  __shared__ half_t lsB[2][128 * LDSPITCH];

  int z  = blockIdx.z;
  int bb = z / p.NHb, hh = z % p.NHb;
  const half_t* A = p.A + (long)bb * p.sAb + (long)hh * p.sAh;
  const half_t* B = p.B + (long)bb * p.sBb + (long)hh * p.sBh;

  int m0 = blockIdx.y * 128;
  int n0 = blockIdx.x * 128;
  int t  = threadIdx.x;
  int wid = t >> 5, lane = t & 31;
  int lrow = lane & 15;
  int hihalf = (lane >> 4) & 1;
  int wm = wid & 3, wn = wid >> 2;

  v8f acc[2][4];
  #pragma unroll
  for (int mt = 0; mt < 2; ++mt)
    #pragma unroll
    for (int nt = 0; nt < 4; ++nt)
      acc[mt][nt] = (v8f){0,0,0,0,0,0,0,0};

  auto stage = [&](int k0, int buf) {
    // A tile 128x32: 512 b128 segments, 2 per thread, async DMA into LDS.
    #pragma unroll
    for (int i = 0; i < 2; i++) {
      int id = i * 256 + t;
      int r = id >> 2, seg = id & 3;
      async_load_b128(&lsA[buf][r * LDSPITCH + seg * 8],
                      A + (long)(m0 + r) * p.lda + k0 + seg * 8);
    }
    if (BNT) {
      // B^T tile 128x32 direct (memory already [N,K]).
      #pragma unroll
      for (int i = 0; i < 2; i++) {
        int id = i * 256 + t;
        int r = id >> 2, seg = id & 3;
        async_load_b128(&lsB[buf][r * LDSPITCH + seg * 8],
                        B + (long)(n0 + r) * p.ldb + k0 + seg * 8);
      }
    } else {
      // B is [K,N]: coalesced read along N, transpose into LDS.
      #pragma unroll
      for (int i = 0; i < 4; i++) {
        int id = i * 256 + t;          // 0..1023
        int k  = id >> 5;              // 0..31
        int ng = id & 31;              // group of 4 columns
        const half_t* src = B + (long)(k0 + k) * p.ldb + n0 + ng * 4;
        half_t e0 = src[0], e1 = src[1], e2 = src[2], e3 = src[3];
        lsB[buf][(ng * 4 + 0) * LDSPITCH + k] = e0;
        lsB[buf][(ng * 4 + 1) * LDSPITCH + k] = e1;
        lsB[buf][(ng * 4 + 2) * LDSPITCH + k] = e2;
        lsB[buf][(ng * 4 + 3) * LDSPITCH + k] = e3;
      }
    }
  };

  stage(0, 0);
  wait_async();
  __syncthreads();

  int buf = 0;
  for (int k0 = 0; k0 < p.K; k0 += 32) {
    if (k0 + 32 < p.K) stage(k0 + 32, buf ^ 1);

    // A fragments, ISA 16-bit 16x32 layout:
    // lanes 0-15 (row=lrow): K 0..7 / 16..23; lanes 16-31: K 8..15 / 24..31
    v16h af[2];
    #pragma unroll
    for (int mt = 0; mt < 2; ++mt) {
      const half_t* ar = &lsA[buf][(wm * 32 + mt * 16 + lrow) * LDSPITCH];
      v8h alo = *(const v8h*)(ar + hihalf * 8);
      v8h ahi = *(const v8h*)(ar + 16 + hihalf * 8);
      #pragma unroll
      for (int i = 0; i < 8; i++) { af[mt][i] = alo[i]; af[mt][i + 8] = ahi[i]; }
    }
    #pragma unroll
    for (int nt = 0; nt < 4; ++nt) {
      const half_t* br = &lsB[buf][(wn * 64 + nt * 16 + lrow) * LDSPITCH];
      v8h blo = *(const v8h*)(br + hihalf * 8);
      v8h bhi = *(const v8h*)(br + 16 + hihalf * 8);
      v16h bf;
      #pragma unroll
      for (int i = 0; i < 8; i++) { bf[i] = blo[i]; bf[i + 8] = bhi[i]; }
      acc[0][nt] = __builtin_amdgcn_wmma_f32_16x16x32_f16(
          false, af[0], false, bf, (short)0, acc[0][nt], false, false);
      acc[1][nt] = __builtin_amdgcn_wmma_f32_16x16x32_f16(
          false, af[1], false, bf, (short)0, acc[1][nt], false, false);
    }

    wait_async();       // prefetched chunk resident before anyone reads it
    __syncthreads();    // all reads of 'buf' done before it is re-staged
    buf ^= 1;
  }

  // ---- epilogue: bias / relu / residual, f32 and/or f16 stores ----
  long cbase = (long)bb * p.sCb + (long)hh * p.sCh;
  long rbase = (long)bb * p.sRb + (long)hh * p.sRh;
  #pragma unroll
  for (int mt = 0; mt < 2; ++mt) {
    #pragma unroll
    for (int nt = 0; nt < 4; ++nt) {
      int ccol = n0 + wn * 64 + nt * 16 + lrow;
      float bi = p.bias ? p.bias[ccol] : 0.0f;
      #pragma unroll
      for (int j = 0; j < 8; j++) {
        // C/D layout: VGPR j -> row j (lanes 0-15) / row 8+j (lanes 16-31)
        int crow = m0 + wm * 32 + mt * 16 + hihalf * 8 + j;
        float v = acc[mt][nt][j] + bi;
        if (p.relu) v = fmaxf(v, 0.0f);
        if (p.R) v += p.R[rbase + (long)crow * p.ldr + ccol];
        long off = cbase + (long)crow * p.ldc + ccol;
        if (p.Cf) p.Cf[off] = v;
        if (p.Ch) p.Ch[off] = (half_t)v;
      }
    }
  }
}

// ---------------------------------------------------------------------------
// Host side
// ---------------------------------------------------------------------------
static void launch_gemm(bool bnt,
                        const half_t* A, long lda, long sAb, long sAh,
                        const half_t* B, long ldb, long sBb, long sBh,
                        float* Cf, half_t* Ch, long ldc, long sCb, long sCh,
                        const float* bias,
                        const float* R, long ldr, long sRb, long sRh,
                        int M, int N, int K, int nh, int nbatch, int relu,
                        hipStream_t stream)
{
  GemmParams p;
  p.A = A; p.lda = lda; p.sAb = sAb; p.sAh = sAh;
  p.B = B; p.ldb = ldb; p.sBb = sBb; p.sBh = sBh;
  p.Cf = Cf; p.Ch = Ch; p.ldc = ldc; p.sCb = sCb; p.sCh = sCh;
  p.bias = bias;
  p.R = R; p.ldr = ldr; p.sRb = sRb; p.sRh = sRh;
  p.M = M; p.N = N; p.K = K; p.NHb = nh; p.relu = relu;
  dim3 grid(N / 128, M / 128, nbatch);
  if (bnt) gemm_wmma_kernel<true><<<grid, dim3(256), 0, stream>>>(p);
  else     gemm_wmma_kernel<false><<<grid, dim3(256), 0, stream>>>(p);
}

// params pytree flattened with sorted dict keys (jax tree order):
// attn{k1,k2,proj1,proj2,q1,q2,v1,v2}{b,w}, attn1{k,proj,q,v}{b,w},
// attn2{...}, d_pos, ln0{b,g}..ln_f{b,g}, mlp{fc1,fc2}{b,w}, x_pos.
enum {
  I_X = 0, I_D = 1,
  I_A_K1B = 2,  I_A_K1W, I_A_K2B, I_A_K2W, I_A_P1B, I_A_P1W, I_A_P2B, I_A_P2W,
  I_A_Q1B, I_A_Q1W, I_A_Q2B, I_A_Q2W, I_A_V1B, I_A_V1W, I_A_V2B, I_A_V2W,
  I_1_KB = 18, I_1_KW, I_1_PB, I_1_PW, I_1_QB, I_1_QW, I_1_VB, I_1_VW,
  I_2_KB = 26, I_2_KW, I_2_PB, I_2_PW, I_2_QB, I_2_QW, I_2_VB, I_2_VW,
  I_DPOS = 34,
  I_LN0B = 35, I_LN0G, I_LN1B, I_LN1G, I_LN2B, I_LN2G, I_LNFB, I_LNFG,
  I_FC1B = 43, I_FC1W, I_FC2B, I_FC2W,
  I_XPOS = 47
};

extern "C" void kernel_launch(void* const* d_in, const int* in_sizes, int n_in,
                              void* d_out, int out_size, void* d_ws, size_t ws_size,
                              hipStream_t stream)
{
  (void)in_sizes; (void)n_in; (void)out_size; (void)ws_size;
  const long BTC = (long)BB * TDIM * CDIM;         // 4,194,304
  const long BTE = (long)BB * TDIM * EDIM;         // 16,777,216
  const long ZTT = (long)BB * NHD * TDIM * TDIM;   // 8,388,608

  char* base = (char*)d_ws;
  size_t off = 0;
  auto alloc = [&](size_t bytes) -> void* {
    void* r = base + off;
    off += (bytes + 255) & ~(size_t)255;
    return r;
  };

  // f32 activation buffers
  float* xt_f  = (float*)alloc(BTC * 4);
  float* dt_f  = (float*)alloc(BTC * 4);
  float* x1_f  = (float*)alloc(BTC * 4);
  float* d1_f  = (float*)alloc(BTC * 4);
  float* xc_f  = (float*)alloc(BTC * 4);
  float* x2_f  = (float*)alloc(BTC * 4);
  float* xo_f  = (float*)alloc(BTC * 4);
  float* sco   = (float*)alloc(ZTT * 4);
  // f16 activation buffers
  half_t* xt_h  = (half_t*)alloc(BTC * 2);
  half_t* dt_h  = (half_t*)alloc(BTC * 2);
  half_t* qh    = (half_t*)alloc(BTC * 2);
  half_t* kh    = (half_t*)alloc(BTC * 2);
  half_t* vh    = (half_t*)alloc(BTC * 2);
  half_t* q2h   = (half_t*)alloc(BTC * 2);
  half_t* k2h   = (half_t*)alloc(BTC * 2);
  half_t* v2h   = (half_t*)alloc(BTC * 2);
  half_t* yh    = (half_t*)alloc(BTC * 2);
  half_t* x1n_h = (half_t*)alloc(BTC * 2);
  half_t* d1n_h = (half_t*)alloc(BTC * 2);
  half_t* h_h   = (half_t*)alloc(BTC * 2);
  half_t* hid_h = (half_t*)alloc(BTE * 2);
  half_t* Ph    = (half_t*)alloc(ZTT * 2);
  // f16 transposed weights [N,K]
  const int NW = 18;
  struct { int idx; int K; int N; half_t* dst; } wd[NW] = {
    { I_1_QW, 512, 512, nullptr }, { I_1_KW, 512, 512, nullptr },
    { I_1_VW, 512, 512, nullptr }, { I_1_PW, 512, 512, nullptr },
    { I_2_QW, 512, 512, nullptr }, { I_2_KW, 512, 512, nullptr },
    { I_2_VW, 512, 512, nullptr }, { I_2_PW, 512, 512, nullptr },
    { I_A_Q1W, 512, 512, nullptr }, { I_A_K1W, 512, 512, nullptr },
    { I_A_V1W, 512, 512, nullptr }, { I_A_P1W, 512, 512, nullptr },
    { I_A_Q2W, 512, 512, nullptr }, { I_A_K2W, 512, 512, nullptr },
    { I_A_V2W, 512, 512, nullptr }, { I_A_P2W, 512, 512, nullptr },
    { I_FC1W, 512, 2048, nullptr }, { I_FC2W, 2048, 512, nullptr },
  };
  for (int i = 0; i < NW; i++)
    wd[i].dst = (half_t*)alloc((size_t)wd[i].K * wd[i].N * 2);
  auto WT = [&](int leaf) -> half_t* {
    for (int i = 0; i < NW; i++) if (wd[i].idx == leaf) return wd[i].dst;
    return nullptr;
  };
  auto F32 = [&](int leaf) -> const float* { return (const float*)d_in[leaf]; };

  // ---- weight prep ----
  for (int i = 0; i < NW; i++) {
    long tot = (long)wd[i].K * wd[i].N;
    wprep_kernel<<<dim3((tot + 255) / 256), dim3(256), 0, stream>>>(
        F32(wd[i].idx), wd[i].dst, wd[i].K, wd[i].N);
  }

  // ---- downsample + positional embedding ----
  {
    dim3 g((unsigned)((BTC + 255) / 256));
    resize_down_kernel<<<g, dim3(256), 0, stream>>>(F32(I_X), F32(I_XPOS), xt_f, xt_h);
    resize_down_kernel<<<g, dim3(256), 0, stream>>>(F32(I_D), F32(I_DPOS), dt_f, dt_h);
  }

  const int M = BB * TDIM;          // 8192
  const int Z = BB * NHD;           // 128
  const long sTC = (long)TDIM * CDIM;
  const float attn_scale = 1.0f / sqrtf((float)DHD);

  // One multi-head self-attention: out_f = res + selfattn(src)
  auto self_attn = [&](const half_t* src_h, const float* res_f, float* out_f,
                       half_t* Wq, const float* bq, half_t* Wk, const float* bk,
                       half_t* Wv, const float* bv, half_t* Wp, const float* bp) {
    launch_gemm(true, src_h, CDIM,0,0, Wq, CDIM,0,0, nullptr, qh, CDIM,0,0,
                bq, nullptr,0,0,0, M, CDIM, CDIM, 1, 1, 0, stream);
    launch_gemm(true, src_h, CDIM,0,0, Wk, CDIM,0,0, nullptr, kh, CDIM,0,0,
                bk, nullptr,0,0,0, M, CDIM, CDIM, 1, 1, 0, stream);
    launch_gemm(true, src_h, CDIM,0,0, Wv, CDIM,0,0, nullptr, vh, CDIM,0,0,
                bv, nullptr,0,0,0, M, CDIM, CDIM, 1, 1, 0, stream);
    // S[z] = q_bh @ k_bh^T  (B memory is [tokens, Dh] == [N,K])
    launch_gemm(true, qh, CDIM, sTC, DHD, kh, CDIM, sTC, DHD,
                sco, nullptr, TDIM, (long)NHD*TDIM*TDIM, (long)TDIM*TDIM,
                nullptr, nullptr,0,0,0, TDIM, TDIM, DHD, NHD, Z, 0, stream);
    softmax_kernel<<<dim3((unsigned)(Z * TDIM / 8)), dim3(256), 0, stream>>>(sco, Ph, attn_scale);
    // y = P @ v  (B memory is [K=tokens, N=Dh], transpose-staged)
    launch_gemm(false, Ph, TDIM, (long)NHD*TDIM*TDIM, (long)TDIM*TDIM,
                vh, CDIM, sTC, DHD, nullptr, yh, CDIM, sTC, DHD,
                nullptr, nullptr,0,0,0, TDIM, DHD, TDIM, NHD, Z, 0, stream);
    // out = res + proj(y)
    launch_gemm(true, yh, CDIM,0,0, Wp, CDIM,0,0, out_f, nullptr, CDIM,0,0,
                bp, res_f, CDIM,0,0, M, CDIM, CDIM, 1, 1, 0, stream);
  };

  // ---- attn1 / attn2 (self-attention streams) ----
  self_attn(xt_h, xt_f, x1_f, WT(I_1_QW), F32(I_1_QB), WT(I_1_KW), F32(I_1_KB),
            WT(I_1_VW), F32(I_1_VB), WT(I_1_PW), F32(I_1_PB));
  self_attn(dt_h, dt_f, d1_f, WT(I_2_QW), F32(I_2_QB), WT(I_2_KW), F32(I_2_KB),
            WT(I_2_VW), F32(I_2_VB), WT(I_2_PW), F32(I_2_PB));

  // ---- layernorms feeding the cross attention ----
  layernorm_kernel<<<dim3(M), dim3(256), 0, stream>>>(x1_f, F32(I_LN1G), F32(I_LN1B), nullptr, x1n_h);
  layernorm_kernel<<<dim3(M), dim3(256), 0, stream>>>(d1_f, F32(I_LN0G), F32(I_LN0B), nullptr, d1n_h);

  // ---- cross attention projections ----
  launch_gemm(true, x1n_h, CDIM,0,0, WT(I_A_Q1W), CDIM,0,0, nullptr, qh, CDIM,0,0,
              F32(I_A_Q1B), nullptr,0,0,0, M, CDIM, CDIM, 1, 1, 0, stream);
  launch_gemm(true, x1n_h, CDIM,0,0, WT(I_A_K1W), CDIM,0,0, nullptr, kh, CDIM,0,0,
              F32(I_A_K1B), nullptr,0,0,0, M, CDIM, CDIM, 1, 1, 0, stream);
  launch_gemm(true, x1n_h, CDIM,0,0, WT(I_A_V1W), CDIM,0,0, nullptr, vh, CDIM,0,0,
              F32(I_A_V1B), nullptr,0,0,0, M, CDIM, CDIM, 1, 1, 0, stream);
  launch_gemm(true, d1n_h, CDIM,0,0, WT(I_A_Q2W), CDIM,0,0, nullptr, q2h, CDIM,0,0,
              F32(I_A_Q2B), nullptr,0,0,0, M, CDIM, CDIM, 1, 1, 0, stream);
  launch_gemm(true, d1n_h, CDIM,0,0, WT(I_A_K2W), CDIM,0,0, nullptr, k2h, CDIM,0,0,
              F32(I_A_K2B), nullptr,0,0,0, M, CDIM, CDIM, 1, 1, 0, stream);
  launch_gemm(true, d1n_h, CDIM,0,0, WT(I_A_V2W), CDIM,0,0, nullptr, v2h, CDIM,0,0,
              F32(I_A_V2B), nullptr,0,0,0, M, CDIM, CDIM, 1, 1, 0, stream);

  // y1 = proj1(attend(q2, k1, v1)); xc = xt + y1
  launch_gemm(true, q2h, CDIM, sTC, DHD, kh, CDIM, sTC, DHD,
              sco, nullptr, TDIM, (long)NHD*TDIM*TDIM, (long)TDIM*TDIM,
              nullptr, nullptr,0,0,0, TDIM, TDIM, DHD, NHD, Z, 0, stream);
  softmax_kernel<<<dim3((unsigned)(Z * TDIM / 8)), dim3(256), 0, stream>>>(sco, Ph, attn_scale);
  launch_gemm(false, Ph, TDIM, (long)NHD*TDIM*TDIM, (long)TDIM*TDIM,
              vh, CDIM, sTC, DHD, nullptr, yh, CDIM, sTC, DHD,
              nullptr, nullptr,0,0,0, TDIM, DHD, TDIM, NHD, Z, 0, stream);
  launch_gemm(true, yh, CDIM,0,0, WT(I_A_P1W), CDIM,0,0, xc_f, nullptr, CDIM,0,0,
              F32(I_A_P1B), xt_f, CDIM,0,0, M, CDIM, CDIM, 1, 1, 0, stream);

  // y2 = proj2(attend(q1, k2, v2)); xc += y2 (residual = xc, in-place safe)
  launch_gemm(true, qh, CDIM, sTC, DHD, k2h, CDIM, sTC, DHD,
              sco, nullptr, TDIM, (long)NHD*TDIM*TDIM, (long)TDIM*TDIM,
              nullptr, nullptr,0,0,0, TDIM, TDIM, DHD, NHD, Z, 0, stream);
  softmax_kernel<<<dim3((unsigned)(Z * TDIM / 8)), dim3(256), 0, stream>>>(sco, Ph, attn_scale);
  launch_gemm(false, Ph, TDIM, (long)NHD*TDIM*TDIM, (long)TDIM*TDIM,
              v2h, CDIM, sTC, DHD, nullptr, yh, CDIM, sTC, DHD,
              nullptr, nullptr,0,0,0, TDIM, DHD, TDIM, NHD, Z, 0, stream);
  launch_gemm(true, yh, CDIM,0,0, WT(I_A_P2W), CDIM,0,0, xc_f, nullptr, CDIM,0,0,
              F32(I_A_P2B), xc_f, CDIM,0,0, M, CDIM, CDIM, 1, 1, 0, stream);

  // ---- MLP: x2 = xc + fc2(relu(fc1(ln2(xc)))) ----
  layernorm_kernel<<<dim3(M), dim3(256), 0, stream>>>(xc_f, F32(I_LN2G), F32(I_LN2B), nullptr, h_h);
  launch_gemm(true, h_h, CDIM,0,0, WT(I_FC1W), CDIM,0,0, nullptr, hid_h, EDIM,0,0,
              F32(I_FC1B), nullptr,0,0,0, M, EDIM, CDIM, 1, 1, 1, stream);
  launch_gemm(true, hid_h, EDIM,0,0, WT(I_FC2W), EDIM,0,0, x2_f, nullptr, CDIM,0,0,
              F32(I_FC2B), xc_f, CDIM,0,0, M, CDIM, EDIM, 1, 1, 0, stream);

  // ---- final layernorm + upsample ----
  layernorm_kernel<<<dim3(M), dim3(256), 0, stream>>>(x2_f, F32(I_LNFG), F32(I_LNFB), xo_f, nullptr);
  {
    long tot = (long)BB * CDIM * HIN * WIN;
    resize_up_kernel<<<dim3((unsigned)((tot + 255) / 256)), dim3(256), 0, stream>>>(
        xo_f, (float*)d_out);
  }
}
